// FeatureDecoder_41832981463439
// MI455X (gfx1250) — compile-verified, matrix operand order checked by reading, not run on hardware
//
#include <hip/hip_runtime.h>
#include <hip/hip_bf16.h>

// MI455X / gfx1250 fused FeatureDecoder, round 2.
// - All heavy GEMMs on v_wmma_f32_16x16x32_f16 (f32 accumulate).
// - Weights repacked once to f16 in exact B-fragment lane order; the packed
//   array is a linear sequence of 17 x 64KB "halves" (c2 + 8 layers x 2).
// - 8 waves / 128 tokens per block; 200KB dynamic LDS:
//   activations (f16, [token][k]) + double-buffered 64KB weight halves.
// - Weight halves staged by the Tensor Data Mover (tensor_load_to_lds,
//   TENSORcnt double-buffer pipeline) when the builtin exists; cooperative
//   b128 copy fallback otherwise.

typedef __attribute__((ext_vector_type(16))) _Float16 v16h;
typedef __attribute__((ext_vector_type(8)))  float    v8f;
typedef __attribute__((ext_vector_type(4)))  unsigned tdm_v4u;
typedef __attribute__((ext_vector_type(8)))  int      tdm_v8i;
typedef __attribute__((ext_vector_type(4)))  int      tdm_v4i;

#if defined(__AMDGCN__) && __has_builtin(__builtin_amdgcn_tensor_load_to_lds) && \
    __has_builtin(__builtin_amdgcn_s_wait_tensorcnt)
#define USE_TDM 1
#else
#define USE_TDM 0
#endif

#define ACT_STRIDE 264                    // 256 + 8 halves pad
#define ACT_BYTES  (128 * ACT_STRIDE * 2) // 67584
#define WBUF_BYTES (2 * 32768 * 2)        // 131072 (two 64KB halves)
#define TINFO_OFF  (ACT_BYTES + WBUF_BYTES)
#define SMEM_BYTES (TINFO_OFF + 128 * 12 * 4)  // 204800

struct KArgs { const float* bias[8]; };

// ---- packed B-fragment index (V_WMMA_F32_16X16X32_F16 B layout, wave32) ----
// lane l holds column n = l&15; kbase = 8*(l>>4);
// VGPR v (0..7) holds K = 32*kc + 16*(v>=4) + kbase + 2*(v&3) and K+1.
__device__ __host__ __forceinline__ int packed_half_index(int k, int n) {
  int kc  = k >> 5;
  int kin = k & 31;
  int nt  = n >> 4;
  int nin = n & 15;
  int lane = nin + ((kin & 8) ? 16 : 0);
  int v    = ((kin & 16) ? 4 : 0) + ((kin >> 1) & 3);
  int h    = v * 2 + (kin & 1);
  return ((kc * 16 + nt) * 32 + lane) * 16 + h;
}

// ---------------- weight repack: fp32 row-major -> f16 packed ---------------
__global__ void repack_w(const float* __restrict__ src,
                         _Float16* __restrict__ dst, int K) {
  int idx = blockIdx.x * 256 + threadIdx.x;
  if (idx >= K * 256) return;
  int k = idx >> 8;
  int n = idx & 255;
  dst[packed_half_index(k, n)] = (_Float16)src[idx];
}

// ---------------- c1 = max over H*W per (b, ch) -----------------------------
__global__ void cmax_kernel(const float* __restrict__ c, float* __restrict__ c1) {
  __shared__ float red[256];
  int bc = blockIdx.x;
  const float* src = c + (size_t)bc * 4096;
  float m = -3.4e38f;
  for (int i = threadIdx.x; i < 4096; i += 256) m = fmaxf(m, src[i]);
  red[threadIdx.x] = m;
  __syncthreads();
  for (int s = 128; s > 0; s >>= 1) {
    if (threadIdx.x < s) red[threadIdx.x] = fmaxf(red[threadIdx.x], red[threadIdx.x + s]);
    __syncthreads();
  }
  if (threadIdx.x == 0) c1[bc] = red[0];
}

// ---------------- per-batch A (3x3), b (3), base[256] -----------------------
__global__ void batch_pre(const float* __restrict__ c1, const float* __restrict__ z,
                          const float* __restrict__ wA,  const float* __restrict__ bA,
                          const float* __restrict__ wb_, const float* __restrict__ bb_,
                          const float* __restrict__ wz,  const float* __restrict__ bz,
                          const float* __restrict__ bp1, const float* __restrict__ bp2,
                          const float* __restrict__ wc1, const float* __restrict__ bc1,
                          const float* __restrict__ bc2,
                          float* __restrict__ pb) {
  __shared__ float c1s[128], zs[128];
  int b = blockIdx.x, tid = threadIdx.x;
  if (tid < 128) { c1s[tid] = c1[b * 128 + tid]; zs[tid] = z[b * 128 + tid]; }
  __syncthreads();
  {
    int n = tid;
    float acc = bz[n] + bp1[n] + bp2[n] + bc1[n] + bc2[n];
    for (int k = 0; k < 128; ++k)
      acc += zs[k] * wz[k * 256 + n] + c1s[k] * wc1[k * 256 + n];
    pb[b * 272 + 12 + n] = acc;
  }
  if (tid < 9) {
    float a = bA[tid];
    for (int k = 0; k < 128; ++k) a += c1s[k] * wA[k * 9 + tid];
    pb[b * 272 + tid] = a;
  } else if (tid < 12) {
    int e = tid - 9;
    float a = bb_[e];
    for (int k = 0; k < 128; ++k) a += c1s[k] * wb_[k * 3 + e];
    pb[b * 272 + tid] = a;
  }
}

// ---------------- fused decoder ---------------------------------------------
__global__ __launch_bounds__(256)
void fused_decoder(const float* __restrict__ p,
                   const float* __restrict__ c,
                   const float* __restrict__ pb,
                   const _Float16* __restrict__ wpk,
                   const float* __restrict__ wp1,
                   const float* __restrict__ wp2,
                   const float* __restrict__ wout,
                   const float* __restrict__ bout,
                   KArgs bz,
                   float* __restrict__ out) {
  extern __shared__ char smem[];
  _Float16* actS = (_Float16*)smem;                       // [128][ACT_STRIDE] f16
  _Float16* wS   = (_Float16*)(smem + ACT_BYTES);         // 2 x 32768 halves
  float (*tinfo)[12] = (float(*)[12])(smem + TINFO_OFF);  // [128][12]

  const int tid   = threadIdx.x;
  const int b     = blockIdx.y;
  const int tile0 = blockIdx.x * 128;
  const int lane  = tid & 31;
  const int wv    = tid >> 5;          // wave 0..7
  const int nin   = lane & 15;
  const int hi    = lane >> 4;
  const int wtok  = wv * 16;

  // ---- TDM staging helpers (1-D tile: 8192 x 8B = one 64KB weight half) ----
  auto stage_issue = [&](int h, int buf) {
#if USE_TDM
    if (wv == 0) {
      unsigned lds_off = (unsigned)(size_t)(wS + ((size_t)buf << 15));
      unsigned long long ga =
          (unsigned long long)(size_t)(wpk + ((size_t)h << 15));
      tdm_v4u g0 = { 1u,                                   // count=1 (valid D#)
                     lds_off,                              // lds_addr
                     (unsigned)ga,                         // global_addr[31:0]
                     (unsigned)((ga >> 32) & 0x01ffffffull) | 0x80000000u }; // type=2
      tdm_v8i g1 = { (int)(3u << 16),      // data_size = 8B
                     (int)(8192u << 16),   // tensor_dim0 = 8192 (lo16)
                     (int)(1u << 16),      // tensor_dim0 hi=0 | tensor_dim1 = 1
                     (int)(8192u << 16),   // tensor_dim1 hi=0 | tile_dim0 = 8192
                     0,                    // tile_dim1 = 0 (1-D), tile_dim2 = 0
                     8192, 0,              // tensor_dim0_stride = 8192
                     0 };
      tdm_v4i gz = { 0, 0, 0, 0 };
#if __clang_major__ >= 23
      tdm_v8i gz8 = { 0, 0, 0, 0, 0, 0, 0, 0 };
      __builtin_amdgcn_tensor_load_to_lds(g0, g1, gz, gz, gz8, 0);
#else
      __builtin_amdgcn_tensor_load_to_lds(g0, g1, gz, gz, 0);
#endif
    }
#else
    (void)h; (void)buf;
#endif
  };

  auto stage_wait = [&](int h, int buf, bool more) {
#if USE_TDM
    (void)h; (void)buf;
    if (wv == 0) {
      if (more) __builtin_amdgcn_s_wait_tensorcnt(1);
      else      __builtin_amdgcn_s_wait_tensorcnt(0);
    }
    __syncthreads();
#else
    (void)more;
    __syncthreads();
    const uint4* s4 = (const uint4*)(wpk + ((size_t)h << 15));
    uint4* d4 = (uint4*)(wS + ((size_t)buf << 15));
    for (int i = tid; i < 4096; i += 256) d4[i] = s4[i];
    __syncthreads();
#endif
  };

  // kick off the first two weight halves immediately
  stage_issue(0, 0);
  stage_issue(1, 1);

  // ---- per-token setup: Ap = p@A + b, perspective, grid coords ----
  if (tid < 128) {
    int t = tid;
    const float* pp = p + ((size_t)b * 4096 + tile0 + t) * 3;
    float p0 = pp[0], p1 = pp[1], p2 = pp[2];
    const float* A = pb + b * 272;
    float ap[3];
#pragma unroll
    for (int e = 0; e < 3; ++e)
      ap[e] = A[9 + e] + p0 * A[e] + p1 * A[3 + e] + p2 * A[6 + e];
    float den = fabsf(ap[2]) + 1e-5f;
    float gx = 2.f * ap[0] / den, gy = 2.f * ap[1] / den;
    float x = (gx + 1.f) * 0.5f * 63.f;
    float y = (gy + 1.f) * 0.5f * 63.f;
    float x0f = floorf(x), y0f = floorf(y);
    int x0 = min(max((int)x0f, 0), 63), x1 = min(max((int)x0f + 1, 0), 63);
    int y0 = min(max((int)y0f, 0), 63), y1 = min(max((int)y0f + 1, 0), 63);
    tinfo[t][0] = p0;   tinfo[t][1] = p1;   tinfo[t][2] = p2;
    tinfo[t][3] = ap[0]; tinfo[t][4] = ap[1]; tinfo[t][5] = ap[2];
    tinfo[t][6] = x - x0f; tinfo[t][7] = y - y0f;
    tinfo[t][8] = (float)x0; tinfo[t][9] = (float)x1;
    tinfo[t][10] = (float)y0; tinfo[t][11] = (float)y1;
  }
  __syncthreads();

  // ---- bilinear border gather: c2[t][ch] -> actS (f16), K = 128 ----
  const float* cb = c + (size_t)b * 128 * 4096;
  for (int i = tid; i < 128 * 128; i += 256) {
    int t = i & 127, ch = i >> 7;
    const float* ti = tinfo[t];
    int x0 = (int)ti[8], x1 = (int)ti[9], y0 = (int)ti[10], y1 = (int)ti[11];
    float wx = ti[6], wy = ti[7];
    const float* pl = cb + ch * 4096;
    float v00 = pl[y0 * 64 + x0], v01 = pl[y0 * 64 + x1];
    float v10 = pl[y1 * 64 + x0], v11 = pl[y1 * 64 + x1];
    float top = v00 + wx * (v01 - v00);
    float bot = v10 + wx * (v11 - v10);
    actS[t * ACT_STRIDE + ch] = (_Float16)(top + wy * (bot - top));
  }

  // ---- init running hidden state: base + p@Wp1 + Ap@Wp2 (D-layout) ----
  v8f x_acc[16];
  const float* baseb = pb + b * 272 + 12;
#pragma unroll
  for (int j = 0; j < 16; ++j) {
    int n = j * 16 + nin;
    float bn = baseb[n];
    float w10 = wp1[n], w11 = wp1[256 + n], w12 = wp1[512 + n];
    float w20 = wp2[n], w21 = wp2[256 + n], w22 = wp2[512 + n];
#pragma unroll
    for (int r = 0; r < 8; ++r) {
      int t = wtok + r + 8 * hi;
      const float* ti = tinfo[t];
      x_acc[j][r] = bn + ti[0] * w10 + ti[1] * w11 + ti[2] * w12
                       + ti[3] * w20 + ti[4] * w21 + ti[5] * w22;
    }
  }

  // A-frag: row = wtok + (lane&15), kbase = 8*(lane>>4),
  // VGPR v: K = 32*kchunk + 16*(v>=4) + kbase + 2*(v&3), +1.
  auto load_afrag = [&](int kchunk) -> v16h {
    union { v16h v; unsigned u[8]; } f;
    const _Float16* row = actS + (wtok + (lane & 15)) * ACT_STRIDE
                          + kchunk * 32 + (lane >> 4) * 8;
#pragma unroll
    for (int v = 0; v < 8; ++v) {
      int koff = ((v & 4) ? 16 : 0) + (v & 3) * 2;
      f.u[v] = *(const unsigned*)(row + koff);
    }
    return f.v;
  };

  auto store_act = [&](v8f* acc) {   // relu -> f16 -> this wave's own rows
#pragma unroll
    for (int j = 0; j < 16; ++j) {
      int n = j * 16 + nin;
#pragma unroll
      for (int r = 0; r < 8; ++r) {
        int t = wtok + r + 8 * hi;
        float v = acc[j][r];
        actS[t * ACT_STRIDE + n] = (_Float16)(v > 0.f ? v : 0.f);
      }
    }
  };

  auto init_bias = [&](v8f* acc, const float* bias) {
#pragma unroll
    for (int j = 0; j < 16; ++j) {
      float bv = bias[j * 16 + nin];
#pragma unroll
      for (int r = 0; r < 8; ++r) acc[j][r] = bv;
    }
  };

  // ---- half-step 0: c2 @ Wc2 (K = 128) into x_acc ----
  stage_wait(0, 0, true);
  {
    const _Float16* wbuf = wS;
#pragma unroll
    for (int kc = 0; kc < 4; ++kc) {
      v16h af = load_afrag(kc);
#pragma unroll
      for (int j = 0; j < 16; ++j) {
        union { v16h v; uint4 q[2]; } bf;
        const uint4* bp = (const uint4*)(wbuf + ((kc * 16 + j) * 32 + lane) * 16);
        bf.q[0] = bp[0];
        bf.q[1] = bp[1];
        x_acc[j] = __builtin_amdgcn_wmma_f32_16x16x32_f16(
            false, af, false, bf.v, (short)0, x_acc[j], false, false);
      }
    }
  }
  __syncthreads();
  stage_issue(2, 0);

  // ---- half-steps 1..16: 8 resnet fc layers, 2 halves each ----
  v8f tmp[16];
#pragma unroll 1
  for (int h = 1; h < 17; ++h) {
    const int buf  = h & 1;
    const int l    = (h - 1) >> 1;   // fc index 0..7
    const int half = (h - 1) & 1;    // K half: 0 -> K 0..127, 1 -> K 128..255
    if (half == 0) {                 // layer boundary: stage relu(input) act
      if (l & 1) store_act(tmp);     // fc_1 input = relu(mid)
      else       store_act(x_acc);   // fc_0 input = relu(x)
      init_bias(tmp, bz.bias[l]);
    }
    stage_wait(h, buf, h < 16);
    const _Float16* wbuf = wS + ((size_t)buf << 15);
    const int kb = half * 4;
#pragma unroll
    for (int kc = 0; kc < 4; ++kc) {
      v16h af = load_afrag(kb + kc);
#pragma unroll
      for (int j = 0; j < 16; ++j) {
        union { v16h v; uint4 q[2]; } bf;
        const uint4* bp = (const uint4*)(wbuf + ((kc * 16 + j) * 32 + lane) * 16);
        bf.q[0] = bp[0];
        bf.q[1] = bp[1];
        tmp[j] = __builtin_amdgcn_wmma_f32_16x16x32_f16(
            false, af, false, bf.v, (short)0, tmp[j], false, false);
      }
    }
    __syncthreads();                 // all waves done with this buffer
    if (h + 2 < 17) stage_issue(h + 2, buf);
    if (half == 1 && (l & 1)) {      // end of a resnet block: x += dx
#pragma unroll
      for (int j = 0; j < 16; ++j) x_acc[j] += tmp[j];
    }
  }

  // ---- fc_out: relu(net) . wout + bias ----
  store_act(x_acc);
  __syncthreads();
  if (tid < 128) {
    int t = tid;
    float s = bout[0];
    for (int n = 0; n < 256; ++n)
      s += (float)actS[t * ACT_STRIDE + n] * wout[n];
    out[(size_t)b * 4096 + tile0 + t] = s;
  }
}

// ---------------- host ----------------
extern "C" void kernel_launch(void* const* d_in, const int* in_sizes, int n_in,
                              void* d_out, int out_size, void* d_ws, size_t ws_size,
                              hipStream_t stream) {
  (void)in_sizes; (void)n_in; (void)out_size; (void)ws_size;
  const float* p    = (const float*)d_in[0];
  const float* z    = (const float*)d_in[1];
  const float* c    = (const float*)d_in[2];
  const float* wA   = (const float*)d_in[3];
  const float* bA   = (const float*)d_in[4];
  const float* wb_  = (const float*)d_in[5];
  const float* bb_  = (const float*)d_in[6];
  const float* wz   = (const float*)d_in[7];
  const float* bzv  = (const float*)d_in[8];
  const float* wp1  = (const float*)d_in[9];
  const float* bp1  = (const float*)d_in[10];
  const float* wp2  = (const float*)d_in[11];
  const float* bp2  = (const float*)d_in[12];
  const float* wc1  = (const float*)d_in[13];
  const float* bc1  = (const float*)d_in[14];
  const float* wc2  = (const float*)d_in[15];
  const float* bc2  = (const float*)d_in[16];
  const float* wout = (const float*)d_in[17];
  const float* bout = (const float*)d_in[18];

  float* ws_f = (float*)d_ws;
  float* c1w  = ws_f;                // 4096 floats
  float* pbw  = ws_f + 4096;         // 32 * 272 floats
  _Float16* wpk = (_Float16*)((char*)d_ws + 51200);  // 17 x 32768 packed halves
  float* out = (float*)d_out;

  (void)hipFuncSetAttribute((const void*)fused_decoder,
                            hipFuncAttributeMaxDynamicSharedMemorySize,
                            SMEM_BYTES);

  cmax_kernel<<<4096, 256, 0, stream>>>(c, c1w);
  batch_pre<<<32, 256, 0, stream>>>(c1w, z, wA, bA, wb_, bb_, wz, bzv,
                                    bp1, bp2, wc1, bc1, bc2, pbw);
  repack_w<<<(128 * 256 + 255) / 256, 256, 0, stream>>>(wc2, wpk, 128);
  size_t off = 32768;
  for (int i = 0; i < 8; ++i) {
    const float* w = (const float*)d_in[19 + i * 2];
    repack_w<<<(256 * 256 + 255) / 256, 256, 0, stream>>>(w, wpk + off, 256);
    off += 65536;
  }

  KArgs ka;
  for (int i = 0; i < 8; ++i) ka.bias[i] = (const float*)d_in[20 + i * 2];
  dim3 grid(32, 32);
  fused_decoder<<<grid, 256, SMEM_BYTES, stream>>>(p, c, pbw, wpk, wp1, wp2,
                                                   wout, bout, ka, out);
}
